// MultiHeadAttention_35656818491827
// MI455X (gfx1250) — matrix operand order repaired and checked by reading, not compiled, not run on hardware
//
#include <hip/hip_runtime.h>
#include <hip/hip_bf16.h>
#include <math.h>

// ---------------------------------------------------------------------------
// MHA for MI455X (gfx1250, wave32). WMMA f16 (f32 accum); LDS-staged,
// double-buffered tiles moved with CDNA5 async-to-LDS (ASYNCcnt); all
// fragments batch-loaded into distinct register sets so WMMAs issue
// back-to-back behind staggered dscnt waits.
// ---------------------------------------------------------------------------

#define USE_ASYNC_LDS 1

typedef _Float16 v16h __attribute__((ext_vector_type(16)));
typedef float    v8f  __attribute__((ext_vector_type(8)));

union H16 { v16h v; uint4 q[2]; _Float16 h[16]; };

constexpr int Nn    = 2048;
constexpr int Dd    = 1024;
constexpr int HEADS = 16;
constexpr int DK    = 64;
constexpr int DH    = 192;   // 2*DK + DV
constexpr int C3    = 3072;  // HEADS*DH
constexpr int ROWS  = 4096;  // Nn*B

// ---- CDNA5 async global->LDS copy (16B per lane, tracked by ASYNCcnt) -----

__device__ __forceinline__ void async_cp16(void* lds, const void* gptr) {
#if USE_ASYNC_LDS
  unsigned l = (unsigned)(size_t)lds;  // LDS addr = low 32 bits of generic ptr
  asm volatile("global_load_async_to_lds_b128 %0, %1, off"
               :: "v"(l), "v"(gptr) : "memory");
#else
  *(uint4*)lds = *(const uint4*)gptr;
#endif
}

__device__ __forceinline__ void async_wait() {
#if USE_ASYNC_LDS
  asm volatile("s_wait_asynccnt 0x0" ::: "memory");
#endif
}

// ---------------- prep: f32 -> f16 convert / tiled transpose-convert -------

__global__ __launch_bounds__(256) void k_cvt(const float* __restrict__ in,
                                             _Float16* __restrict__ out, int n) {
  int i = blockIdx.x * 256 + threadIdx.x;
  if (i < n) out[i] = (_Float16)in[i];
}

// in: R x C (row-major) f32, out: C x R (row-major) f16. 32x32 LDS tiles.
__global__ __launch_bounds__(256) void k_cvt_t(const float* __restrict__ in,
                                               _Float16* __restrict__ out,
                                               int R, int C) {
  __shared__ _Float16 t[32][33];
  const int c0 = blockIdx.x * 32, k0 = blockIdx.y * 32;
  const int tx = threadIdx.x, ty = threadIdx.y;  // block (32, 8)
#pragma unroll
  for (int i = 0; i < 4; ++i) {
    int k = k0 + ty + i * 8, c = c0 + tx;
    t[ty + i * 8][tx] = (_Float16)in[(size_t)k * C + c];
  }
  __syncthreads();
#pragma unroll
  for (int i = 0; i < 4; ++i) {
    int c = c0 + ty + i * 8, k = k0 + tx;
    out[(size_t)c * R + k] = t[tx][ty + i * 8];
  }
}

// ---------------------------------------------------------------------------
// Shared GEMM core: block = 256 thr (8 waves), tile 128(M) x 64(N), K = 1024
// in 64-K stages. B tile (64 cols x 64 K) staged in LDS via async DMA,
// double-buffered, padded row stride 72 halves (36-dword lane step = bank
// permutation -> conflict-free b128 DS reads). A fragments register
// double-buffered. All 8 B frags loaded into distinct registers, then
// 8 back-to-back WMMAs per barrier pair.
// ---------------------------------------------------------------------------

constexpr int BSTR = 72;  // 64 K halves + 8 pad

__device__ __forceinline__ void gemm_core_128x64(
    const _Float16* __restrict__ Arow,   // A row base for this lane (M0+lm)
    const _Float16* __restrict__ WT,     // (cols x K) row-major = B^T
    _Float16* __restrict__ Bt,           // LDS: 2 * 64*BSTR halves
    int N0, int lm, int lh, int tid, v8f acc[4]) {
  auto stage = [&](int kk, int buf) {    // 512 8-half chunks, 2 per thread
#pragma unroll
    for (int r = 0; r < 2; ++r) {
      int id = tid + r * 256;
      int col = id >> 3, seg = (id & 7) * 8;
      async_cp16(Bt + buf * 64 * BSTR + col * BSTR + seg,
                 WT + (size_t)(N0 + col) * Dd + kk + seg);
    }
  };
  auto loadA = [&](H16& a, int kpos) {
    const _Float16* ap = Arow + kpos + lh * 8;
    a.q[0] = *(const uint4*)(ap);
    a.q[1] = *(const uint4*)(ap + 16);
  };

  stage(0, 0);
  H16 a0, a1;
  loadA(a0, 0);
  loadA(a1, 32);
  async_wait();
  __syncthreads();

  for (int kk = 0; kk < Dd; kk += 64) {
    const int buf = (kk >> 6) & 1;
    if (kk + 64 < Dd) stage(kk + 64, buf ^ 1);      // DMA next stage
    H16 a0n = a0, a1n = a1;
    if (kk + 64 < Dd) { loadA(a0n, kk + 64); loadA(a1n, kk + 96); }

    const _Float16* bb = Bt + buf * 64 * BSTR + lm * BSTR + lh * 8;
    H16 b0[4], b1[4];                    // distinct register sets: all loads
#pragma unroll
    for (int nb = 0; nb < 4; ++nb) {     // issue before any WMMA consumes
      b0[nb].q[0] = *(const uint4*)(bb + nb * 16 * BSTR);
      b0[nb].q[1] = *(const uint4*)(bb + nb * 16 * BSTR + 16);
    }
#pragma unroll
    for (int nb = 0; nb < 4; ++nb) {
      b1[nb].q[0] = *(const uint4*)(bb + nb * 16 * BSTR + 32);
      b1[nb].q[1] = *(const uint4*)(bb + nb * 16 * BSTR + 48);
    }
#pragma unroll
    for (int nb = 0; nb < 4; ++nb)
      acc[nb] = __builtin_amdgcn_wmma_f32_16x16x32_f16(
          false, a0.v, false, b0[nb].v, (short)0, acc[nb], false, false);
#pragma unroll
    for (int nb = 0; nb < 4; ++nb)
      acc[nb] = __builtin_amdgcn_wmma_f32_16x16x32_f16(
          false, a1.v, false, b1[nb].v, (short)0, acc[nb], false, false);

    a0 = a0n; a1 = a1n;
    async_wait();        // next tile landed
    __syncthreads();     // everyone done reading current buffer
  }
}

// ---------------- QKV GEMM + scatter to head-major Q/K/V -------------------

__global__ __launch_bounds__(256) void k_qkv(const _Float16* __restrict__ Sh,
                                             const _Float16* __restrict__ WT,
                                             const float* __restrict__ bias,
                                             _Float16* __restrict__ Qo,
                                             _Float16* __restrict__ Ko,
                                             _Float16* __restrict__ Vo) {
  __shared__ __align__(16) _Float16 Bt[2 * 64 * BSTR];
  const int lane = threadIdx.x & 31;
  const int wave = threadIdx.x >> 5;
  const int lm = lane & 15, lh = lane >> 4;
  const int M0 = blockIdx.y * 128 + wave * 16;
  const int N0 = blockIdx.x * 64;

  v8f acc[4] = {};
  gemm_core_128x64(Sh + (size_t)(M0 + lm) * Dd, WT, Bt, N0, lm, lh,
                   (int)threadIdx.x, acc);

#pragma unroll
  for (int nb = 0; nb < 4; ++nb) {
#pragma unroll
    for (int e = 0; e < 8; ++e) {
      int r = M0 + lh * 8 + e;          // row of (N*B, 3072) qkv matrix
      int c = N0 + nb * 16 + lm;        // column
      float v = acc[nb][e] + bias[c];
      int nrow = r >> 1, bidx = r & 1;  // r = n*2 + b
      int head = c / DH, d = c - head * DH;
      int hh = bidx * HEADS + head;
      _Float16 hv = (_Float16)v;
      size_t base = ((size_t)hh * Nn + nrow) * DK;
      if (d < DK)            Qo[base + d] = hv;
      else if (d < 2 * DK)   Ko[base + (d - DK)] = hv;
      else                   Vo[base + (d - 2 * DK)] = hv;
    }
  }
}

// ---------------- flash attention per head ---------------------------------
// block = 8 waves x 16 query rows = 128 rows, so each 32-key K/V tile is
// reused by 8 waves (2x the previous reuse, half the DMA + barrier count).
// blockIdx.x = head -> the 32 heads of a row tile co-resident, sharing mask
// lines in L2. K tile async-DMA'd; V transposed into LDS at copy time.

constexpr int KSTR = 72;  // K tile LDS row stride (halves)
constexpr int VSTR = 40;  // V^T tile LDS row stride (halves)
constexpr int PSTR = 40;  // P slab row stride (halves)

__global__ __launch_bounds__(256) void k_attn(const _Float16* __restrict__ Qb,
                                              const _Float16* __restrict__ Kb,
                                              const _Float16* __restrict__ Vb,
                                              const float* __restrict__ mask,
                                              _Float16* __restrict__ Vals) {
  __shared__ __align__(16) _Float16 Kt[2 * 32 * KSTR];   //  9.0 KB
  __shared__ __align__(16) _Float16 Vt[2 * 64 * VSTR];   // 10.0 KB
  __shared__ __align__(16) _Float16 Pl[8 * 16 * PSTR];   // 10.0 KB
  const int lane = threadIdx.x & 31;
  const int wave = threadIdx.x >> 5;
  const int lm = lane & 15, lh = lane >> 4;
  const int hh = blockIdx.x;                      // head fastest -> L2 reuse
  const int i0 = blockIdx.y * 128 + wave * 16;

  const _Float16* Qh = Qb + (size_t)hh * Nn * DK;
  const _Float16* Kh = Kb + (size_t)hh * Nn * DK;
  const _Float16* Vh = Vb + (size_t)hh * Nn * DK;

  const int f0 = threadIdx.x;          // 256 chunks = exactly one 32x64 tile
  H16 qa[2];
#pragma unroll
  for (int t = 0; t < 2; ++t) {
    const _Float16* qp = Qh + (size_t)(i0 + lm) * DK + t * 32 + lh * 8;
    qa[t].q[0] = *(const uint4*)(qp);
    qa[t].q[1] = *(const uint4*)(qp + 16);
  }

  auto copy_K = [&](int j0, int buf) {
    int key = f0 >> 3, d8 = (f0 & 7) * 8;
    async_cp16(&Kt[buf * 32 * KSTR + key * KSTR + d8],
               Kh + (size_t)(j0 + key) * DK + d8);
  };
  auto copy_Vt = [&](int j0, int buf) {   // transpose at copy time
    int key = f0 >> 3, d8 = (f0 & 7) * 8;
    H16 x;
    x.q[0] = *(const uint4*)(Vh + (size_t)(j0 + key) * DK + d8);
#pragma unroll
    for (int i = 0; i < 8; ++i)
      Vt[buf * 64 * VSTR + (d8 + i) * VSTR + key] = x.h[i];
  };

  float m8[8], l8[8];
#pragma unroll
  for (int e = 0; e < 8; ++e) { m8[e] = -1e30f; l8[e] = 0.f; }
  v8f o[4] = {};
  const int irow = i0 + lh * 8;

  copy_K(0, 0);
  copy_Vt(0, 0);
  async_wait();
  __syncthreads();

  for (int j0 = 0; j0 < Nn; j0 += 32) {
    const int buf = (j0 >> 5) & 1;
    if (j0 + 32 < Nn) {          // DMA next tile while computing this one
      copy_K(j0 + 32, buf ^ 1);
      copy_Vt(j0 + 32, buf ^ 1);
    }

    // ---- S = Q K^T + mask: load all 4 K frags, then 4 WMMAs -------------
    const _Float16* kpb = &Kt[buf * 32 * KSTR + lm * KSTR + lh * 8];
    H16 kb[4];
#pragma unroll
    for (int u = 0; u < 2; ++u)
#pragma unroll
      for (int t = 0; t < 2; ++t) {
        const _Float16* kp = kpb + u * 16 * KSTR + t * 32;
        kb[u * 2 + t].q[0] = *(const uint4*)(kp);
        kb[u * 2 + t].q[1] = *(const uint4*)(kp + 16);
      }
    v8f s[2] = {};
#pragma unroll
    for (int u = 0; u < 2; ++u) {
      s[u] = __builtin_amdgcn_wmma_f32_16x16x32_f16(
          false, qa[0].v, false, kb[u * 2].v, (short)0, s[u], false, false);
      s[u] = __builtin_amdgcn_wmma_f32_16x16x32_f16(
          false, qa[1].v, false, kb[u * 2 + 1].v, (short)0, s[u], false, false);
    }
#pragma unroll
    for (int u = 0; u < 2; ++u)
#pragma unroll
      for (int e = 0; e < 8; ++e)
        s[u][e] += mask[(size_t)(irow + e) * Nn + (j0 + u * 16 + lm)];

    // ---- online softmax (width-16 xor shuffles, wave32) -----------------
    float alpha[8];
#pragma unroll
    for (int e = 0; e < 8; ++e) {
      float t = fmaxf(s[0][e], s[1][e]);
#pragma unroll
      for (int off = 8; off > 0; off >>= 1)
        t = fmaxf(t, __shfl_xor(t, off, 16));
      float mn = fmaxf(m8[e], t);
      alpha[e] = __expf(m8[e] - mn);
      float p0 = __expf(s[0][e] - mn);
      float p1 = __expf(s[1][e] - mn);
      s[0][e] = p0; s[1][e] = p1;
      float ps = p0 + p1;
#pragma unroll
      for (int off = 8; off > 0; off >>= 1)
        ps += __shfl_xor(ps, off, 16);
      l8[e] = l8[e] * alpha[e] + ps;
      m8[e] = mn;
    }
#pragma unroll
    for (int nb = 0; nb < 4; ++nb)
#pragma unroll
      for (int e = 0; e < 8; ++e) o[nb][e] *= alpha[e];

    // ---- P relayout (C layout -> A frag) via padded per-wave LDS slab ---
#pragma unroll
    for (int u = 0; u < 2; ++u)
#pragma unroll
      for (int e = 0; e < 8; ++e)
        Pl[wave * 16 * PSTR + (lh * 8 + e) * PSTR + u * 16 + lm] =
            (_Float16)s[u][e];
    __syncthreads();
    H16 pa;
    {
      const _Float16* pp = &Pl[wave * 16 * PSTR + lm * PSTR + lh * 8];
      pa.q[0] = *(const uint4*)(pp);
      pa.q[1] = *(const uint4*)(pp + 16);
    }

    // ---- O += P x V: load all 4 V frags, then 4 WMMAs -------------------
    H16 vbf[4];
#pragma unroll
    for (int nb = 0; nb < 4; ++nb) {
      const _Float16* vp = &Vt[buf * 64 * VSTR + (nb * 16 + lm) * VSTR + lh * 8];
      vbf[nb].q[0] = *(const uint4*)(vp);
      vbf[nb].q[1] = *(const uint4*)(vp + 16);
    }
#pragma unroll
    for (int nb = 0; nb < 4; ++nb)
      o[nb] = __builtin_amdgcn_wmma_f32_16x16x32_f16(
          false, pa.v, false, vbf[nb].v, (short)0, o[nb], false, false);

    async_wait();      // next K tile landed
    __syncthreads();   // all waves done with current buffers + V stores visible
  }

  // normalize and store into (N*B, HEADS*DV) layout for the output GEMM
  const int bi = hh >> 4, head = hh & 15;
#pragma unroll
  for (int nb = 0; nb < 4; ++nb)
#pragma unroll
    for (int e = 0; e < 8; ++e) {
      int row = irow + e;
      float v = o[nb][e] / l8[e];
      Vals[((size_t)(row * 2 + bi)) * (HEADS * DK) + head * DK + nb * 16 + lm] =
          (_Float16)v;
    }
}

// ---------------- output GEMM: (4096x1024) x (1024x1024) + bias -> f32 -----

__global__ __launch_bounds__(256) void k_out(const _Float16* __restrict__ Vals,
                                             const _Float16* __restrict__ WT,
                                             const float* __restrict__ bias,
                                             float* __restrict__ out) {
  __shared__ __align__(16) _Float16 Bt[2 * 64 * BSTR];
  const int lane = threadIdx.x & 31;
  const int wave = threadIdx.x >> 5;
  const int lm = lane & 15, lh = lane >> 4;
  const int M0 = blockIdx.y * 128 + wave * 16;
  const int N0 = blockIdx.x * 64;

  v8f acc[4] = {};
  gemm_core_128x64(Vals + (size_t)(M0 + lm) * Dd, WT, Bt, N0, lm, lh,
                   (int)threadIdx.x, acc);

#pragma unroll
  for (int nb = 0; nb < 4; ++nb)
#pragma unroll
    for (int e = 0; e < 8; ++e) {
      int r = M0 + lh * 8 + e, c = N0 + nb * 16 + lm;
      out[(size_t)r * Dd + c] = acc[nb][e] + bias[c];
    }
}

// ---------------------------------------------------------------------------

extern "C" void kernel_launch(void* const* d_in, const int* in_sizes, int n_in,
                              void* d_out, int out_size, void* d_ws, size_t ws_size,
                              hipStream_t stream) {
  const float* stream_in = (const float*)d_in[0];
  const float* mask      = (const float*)d_in[1];
  const float* w_qkv     = (const float*)d_in[2];
  const float* b_qkv     = (const float*)d_in[3];
  const float* w_out     = (const float*)d_in[4];
  const float* b_out     = (const float*)d_in[5];
  float* out = (float*)d_out;

  // workspace layout (48 MB total)
  char* ws = (char*)d_ws;
  _Float16* Sh    = (_Float16*)(ws);                  //  8 MB stream f16
  _Float16* WqkvT = (_Float16*)(ws + (8ull  << 20));  //  6 MB w_qkv^T f16
  _Float16* WoutT = (_Float16*)(ws + (14ull << 20));  //  2 MB w_out^T f16
  _Float16* Qb    = (_Float16*)(ws + (16ull << 20));  //  8 MB Q head-major
  _Float16* Kb    = (_Float16*)(ws + (24ull << 20));  //  8 MB K head-major
  _Float16* Vb    = (_Float16*)(ws + (32ull << 20));  //  8 MB V head-major
  _Float16* Vals  = (_Float16*)(ws + (40ull << 20));  //  8 MB attn out f16

  int n1 = ROWS * Dd;
  k_cvt<<<dim3(n1 / 256), 256, 0, stream>>>(stream_in, Sh, n1);
  k_cvt_t<<<dim3(C3 / 32, Dd / 32), dim3(32, 8), 0, stream>>>(w_qkv, WqkvT,
                                                              Dd, C3);
  k_cvt_t<<<dim3(Dd / 32, Dd / 32), dim3(32, 8), 0, stream>>>(w_out, WoutT,
                                                              Dd, Dd);

  k_qkv<<<dim3(C3 / 64, ROWS / 128), 256, 0, stream>>>(Sh, WqkvT, b_qkv,
                                                       Qb, Kb, Vb);
  k_attn<<<dim3(32, Nn / 128), 256, 0, stream>>>(Qb, Kb, Vb, mask, Vals);
  k_out<<<dim3(Dd / 64, ROWS / 128), 256, 0, stream>>>(Vals, WoutT, b_out, out);
}
